// GCN_4999341932783
// MI455X (gfx1250) — compile-verified, matrix operand order checked by reading, not run on hardware
//
#include <hip/hip_runtime.h>
#include <hip/hip_fp16.h>

typedef __attribute__((ext_vector_type(16))) _Float16 v16h;
typedef __attribute__((ext_vector_type(8)))  float    v8f;

static constexpr int N_NODES = 50000;
static constexpr int N_EDGES = 800000;
static constexpr int F_INP   = 256;
static constexpr int H_DIM   = 64;
static constexpr int N_CLS   = 10;
static constexpr int N_GRAPH = 64;
static constexpr int MT      = 5;     // M tiles per wave: 3125 row tiles = 625 * 5 (exact)
static constexpr int GEMM_TB = 320;   // 10 waves/block; 2500 waves = 250 blocks (exact)

// ---------------- utility ----------------
__global__ void zero_kernel(float* __restrict__ p, int n) {
  int i = blockIdx.x * blockDim.x + threadIdx.x;
  if (i < n) p[i] = 0.0f;
}

// ---------------- degree / normalization ----------------
__global__ void deg_kernel(const int* __restrict__ dst, float* __restrict__ deg, int E) {
  int e = blockIdx.x * blockDim.x + threadIdx.x;
  if (e < E) unsafeAtomicAdd(&deg[dst[e]], 1.0f);
}

__global__ void dis_kernel(const float* __restrict__ deg, float* __restrict__ dis, int n) {
  int i = blockIdx.x * blockDim.x + threadIdx.x;
  if (i < n) dis[i] = rsqrtf(deg[i] + 1.0f);
}

__global__ void edge_norm_kernel(const int* __restrict__ src, const int* __restrict__ dst,
                                 const float* __restrict__ dis, float* __restrict__ en, int E) {
  int e = blockIdx.x * blockDim.x + threadIdx.x;
  if (e < E) en[e] = dis[src[e]] * dis[dst[e]];
}

// ---------------- WMMA GEMM: out[N,64] = A[N,K](f32->f16) @ W[K,64](f32->f16) ----------------
// W is converted to f16 once per block into LDS, pre-swizzled into per-lane WMMA
// B-fragment order ([chunk][tileN][lane][16 halves]) -> one fragment = two ds_load_b128.
// Each wave register-blocks MT=5 row tiles (exact tiling: no bounds checks, EXEC
// stays all-ones, all tile indices scalarized via readfirstlane).
template <int K>
__global__ __launch_bounds__(GEMM_TB) void wmma_gemm_kernel(const float* __restrict__ A,
                                                            const float* __restrict__ W,
                                                            float* __restrict__ out) {
  constexpr int CHUNKS = K / 32;
  __shared__ _Float16 ldsW[CHUNKS * 4 * 32 * 16];   // K * 64 halves

  const int tid = threadIdx.x;
  // cooperative f32 -> f16 swizzled fill of W fragments
  for (int idx = tid; idx < CHUNKS * 4 * 32 * 16; idx += GEMM_TB) {
    int j     = idx & 15;
    int lane  = (idx >> 4) & 31;
    int tn    = (idx >> 9) & 3;
    int chunk = idx >> 11;
    int k = chunk * 32 + ((lane >> 4) << 3) + (j < 8 ? j : j + 8);  // ISA 16-bit K-striping
    int n = tn * 16 + (lane & 15);
    ldsW[idx] = (_Float16)W[k * H_DIM + n];
  }
  __syncthreads();

  // scalar (SGPR) tile coordinates: wave id is uniform by construction
  const int wave  = blockIdx.x * (GEMM_TB / 32) + __builtin_amdgcn_readfirstlane(tid >> 5);
  const int lane  = tid & 31;
  const int tileN = wave & 3;
  const int tile0 = (wave >> 2) * MT;              // first of MT row tiles

  const int half16 = lane >> 4;                    // 0: lanes 0-15, 1: lanes 16-31
  const int hiK    = half16 * 8;
  const int rowSel = lane & 15;
  const int ncol   = tileN * 16 + rowSel;

  v8f acc[MT] = {};
#pragma unroll
  for (int chunk = 0; chunk < CHUNKS; ++chunk) {
    const int kk = chunk * 32;
    v16h b = *(const v16h*)&ldsW[((chunk * 4 + tileN) * 32 + lane) * 16];
#pragma unroll
    for (int t = 0; t < MT; ++t) {
      const float* ap = A + (size_t)((tile0 + t) * 16 + rowSel) * K + kk + hiK;
      float af[16];
      *(float4*)(af + 0)  = *(const float4*)(ap + 0);
      *(float4*)(af + 4)  = *(const float4*)(ap + 4);
      *(float4*)(af + 8)  = *(const float4*)(ap + 16);
      *(float4*)(af + 12) = *(const float4*)(ap + 20);
      v16h a;
#pragma unroll
      for (int j = 0; j < 16; ++j) a[j] = (_Float16)af[j];
      acc[t] = __builtin_amdgcn_wmma_f32_16x16x32_f16(false, a, false, b, (short)0,
                                                      acc[t], false, false);
    }
  }

  // D store: VGPR r -> M = r + 8*half16 (32-bit C/D layout)
#pragma unroll
  for (int t = 0; t < MT; ++t) {
    const int mbase = (tile0 + t) * 16 + hiK;
#pragma unroll
    for (int r = 0; r < 8; ++r)
      out[(size_t)(mbase + r) * H_DIM + ncol] = acc[t][r];
  }
}

// ---------------- edge scatter: agg[dst] += h[src] * en[e] ----------------
__global__ void scatter_kernel(const int* __restrict__ src, const int* __restrict__ dst,
                               const float* __restrict__ en, const float* __restrict__ h,
                               float* __restrict__ agg, int E) {
  int idx = blockIdx.x * blockDim.x + threadIdx.x;   // E * 16 work-items
  int e = idx >> 4;
  if (e >= E) return;
  int f = (idx & 15) << 2;
  float w = en[e];
  float4 v = *(const float4*)(h + (size_t)src[e] * H_DIM + f);
  float* a = agg + (size_t)dst[e] * H_DIM + f;
  unsafeAtomicAdd(a + 0, v.x * w);
  unsafeAtomicAdd(a + 1, v.y * w);
  unsafeAtomicAdd(a + 2, v.z * w);
  unsafeAtomicAdd(a + 3, v.w * w);
}

// ---------------- self-loop + bias + relu ----------------
__global__ void finalize_kernel(const float* __restrict__ agg, const float* __restrict__ hpre,
                                const float* __restrict__ dis, const float* __restrict__ bias,
                                float* __restrict__ out, int n) {
  int idx = blockIdx.x * blockDim.x + threadIdx.x;   // n * 16
  int node = idx >> 4;
  if (node >= n) return;
  int f = (idx & 15) << 2;
  float sn = dis[node]; sn *= sn;                    // self-loop weight deg^-1
  float4 a  = *(const float4*)(agg  + (size_t)node * H_DIM + f);
  float4 hp = *(const float4*)(hpre + (size_t)node * H_DIM + f);
  float4 b  = *(const float4*)(bias + f);
  float4 r;
  r.x = fmaxf(a.x + hp.x * sn + b.x, 0.0f);
  r.y = fmaxf(a.y + hp.y * sn + b.y, 0.0f);
  r.z = fmaxf(a.z + hp.z * sn + b.z, 0.0f);
  r.w = fmaxf(a.w + hp.w * sn + b.w, 0.0f);
  *(float4*)(out + (size_t)node * H_DIM + f) = r;
}

// ---------------- global mean pool (sum part) ----------------
__global__ void pool_kernel(const float* __restrict__ h, const int* __restrict__ batch,
                            float* __restrict__ pooled, int n) {
  int idx = blockIdx.x * blockDim.x + threadIdx.x;   // n * 16
  int node = idx >> 4;
  if (node >= n) return;
  int f = (idx & 15) << 2;
  int g = batch[node];
  float4 v = *(const float4*)(h + (size_t)node * H_DIM + f);
  float* p = pooled + (size_t)g * H_DIM + f;
  unsafeAtomicAdd(p + 0, v.x);
  unsafeAtomicAdd(p + 1, v.y);
  unsafeAtomicAdd(p + 2, v.z);
  unsafeAtomicAdd(p + 3, v.w);
}

__global__ void count_kernel(const int* __restrict__ batch, float* __restrict__ counts, int n) {
  int i = blockIdx.x * blockDim.x + threadIdx.x;
  if (i < n) unsafeAtomicAdd(&counts[batch[i]], 1.0f);
}

// ---------------- final linear: out[g,c] = (pooled[g]/count[g]) @ linW + linb ----------------
__global__ void final_linear_kernel(const float* __restrict__ pooled,
                                    const float* __restrict__ counts,
                                    const float* __restrict__ linW,
                                    const float* __restrict__ linb,
                                    float* __restrict__ out) {
  int t = threadIdx.x;
  if (t >= N_GRAPH * N_CLS) return;
  int g = t / N_CLS, c = t - g * N_CLS;
  float s = 0.0f;
#pragma unroll
  for (int k = 0; k < H_DIM; ++k) s += pooled[g * H_DIM + k] * linW[k * N_CLS + c];
  float cnt = counts[g];
  cnt = cnt < 1.0f ? 1.0f : cnt;
  out[t] = s / cnt + linb[c];
}

extern "C" void kernel_launch(void* const* d_in, const int* in_sizes, int n_in,
                              void* d_out, int out_size, void* d_ws, size_t ws_size,
                              hipStream_t stream) {
  const float* x      = (const float*)d_in[0];
  const int*   ei     = (const int*)d_in[1];     // [2, E] flat
  const int*   batch  = (const int*)d_in[2];
  const float* W1     = (const float*)d_in[3];
  const float* b1     = (const float*)d_in[4];
  const float* W2     = (const float*)d_in[5];
  const float* b2     = (const float*)d_in[6];
  const float* W3     = (const float*)d_in[7];
  const float* b3     = (const float*)d_in[8];
  const float* linW   = (const float*)d_in[9];
  const float* linb   = (const float*)d_in[10];
  float* out = (float*)d_out;

  const int N = N_NODES, E = N_EDGES;
  const int* src = ei;
  const int* dst = ei + E;

  // workspace layout (floats)
  float* ws     = (float*)d_ws;
  float* deg    = ws;                            // N
  float* dis    = deg + N;                       // N
  float* en     = dis + N;                       // E
  float* h0     = en + E;                        // N*H  (pre-agg GEMM result)
  float* h1     = h0 + (size_t)N * H_DIM;        // N*H  (aggregation accumulator)
  float* h2     = h1 + (size_t)N * H_DIM;        // N*H  (layer output)
  float* pooled = h2 + (size_t)N * H_DIM;        // G*H
  float* counts = pooled + N_GRAPH * H_DIM;      // G

  const int TB = 256;
  auto blk = [](long n, int tb) { return (int)((n + tb - 1) / tb); };

  // --- degrees & norms ---
  zero_kernel<<<blk(N, TB), TB, 0, stream>>>(deg, N);
  deg_kernel<<<blk(E, TB), TB, 0, stream>>>(dst, deg, E);
  dis_kernel<<<blk(N, TB), TB, 0, stream>>>(deg, dis, N);
  edge_norm_kernel<<<blk(E, TB), TB, 0, stream>>>(src, dst, dis, en, E);

  // exact GEMM tiling: 3125 row tiles = 625 groups of MT=5; 625*4 = 2500 waves;
  // 2500 waves / 10 waves-per-block = 250 blocks, no remainder anywhere.
  const int gemmBlocks = (N / 16 / MT) * (H_DIM / 16) * 32 / GEMM_TB;   // 250
  const long nh = (long)N * H_DIM;

  // --- layer 1 ---
  wmma_gemm_kernel<F_INP><<<gemmBlocks, GEMM_TB, 0, stream>>>(x, W1, h0);
  zero_kernel<<<blk(nh, TB), TB, 0, stream>>>(h1, (int)nh);
  scatter_kernel<<<blk((long)E * 16, TB), TB, 0, stream>>>(src, dst, en, h0, h1, E);
  finalize_kernel<<<blk((long)N * 16, TB), TB, 0, stream>>>(h1, h0, dis, b1, h2, N);

  // --- layer 2 ---
  wmma_gemm_kernel<H_DIM><<<gemmBlocks, GEMM_TB, 0, stream>>>(h2, W2, h0);
  zero_kernel<<<blk(nh, TB), TB, 0, stream>>>(h1, (int)nh);
  scatter_kernel<<<blk((long)E * 16, TB), TB, 0, stream>>>(src, dst, en, h0, h1, E);
  finalize_kernel<<<blk((long)N * 16, TB), TB, 0, stream>>>(h1, h0, dis, b2, h2, N);

  // --- layer 3 ---
  wmma_gemm_kernel<H_DIM><<<gemmBlocks, GEMM_TB, 0, stream>>>(h2, W3, h0);
  zero_kernel<<<blk(nh, TB), TB, 0, stream>>>(h1, (int)nh);
  scatter_kernel<<<blk((long)E * 16, TB), TB, 0, stream>>>(src, dst, en, h0, h1, E);
  finalize_kernel<<<blk((long)N * 16, TB), TB, 0, stream>>>(h1, h0, dis, b3, h2, N);

  // --- pool + classify ---
  zero_kernel<<<blk(N_GRAPH * H_DIM + N_GRAPH, TB), TB, 0, stream>>>(pooled,
                                                                     N_GRAPH * H_DIM + N_GRAPH);
  pool_kernel<<<blk((long)N * 16, TB), TB, 0, stream>>>(h2, batch, pooled, N);
  count_kernel<<<blk(N, TB), TB, 0, stream>>>(batch, counts, N);
  final_linear_kernel<<<1, 640, 0, stream>>>(pooled, counts, linW, linb, out);
}
